// Conv2D_87058987090635
// MI455X (gfx1250) — matrix operand order repaired and checked by reading, not compiled, not run on hardware
//
#include <hip/hip_runtime.h>
#include <stdint.h>

// ---------------------------------------------------------------------------
// 4096x4096 f32 'VALID' conv with a 16x16 kernel -> 4081x4081 f32.
// Strategy: bf16 hi/lo split-precision WMMA (v_wmma_f32_16x16x32_bf16),
// row-direction taps folded into K=32 via a banded Toeplitz weight matrix.
//   Y_tile(16x16) = sum_u A_u(16x32) x T_u(32x16)
//   A_u[i,k] = X[i0+i+u, j0+k],  T_u[k,j] = W[u, k-j] for 0<=k-j<16 else 0
// Each f32 operand is split x = hi + lo (bf16 each); we accumulate
// hi*hi + hi*lo + lo*hi in f32 (near-f32 accuracy, 3 WMMAs per u per tile).
// ---------------------------------------------------------------------------

typedef __attribute__((ext_vector_type(16))) __bf16    v16bf;
typedef __attribute__((ext_vector_type(8)))  float     v8f;
typedef __attribute__((ext_vector_type(8)))  uint32_t  v8u;

#define HH 4096
#define WW 4096
#define OH 4081
#define OW 4081

#define BT      128          // output tile side per workgroup
#define NTHREADS 256         // 8 wave32
#define TJ      8            // 16x16 tiles across per wave
#define IN_DIM  144          // input patch side (BT + 16)
#define XP      73           // LDS X row stride in u32 (72 bf16-pairs + 1 pad, odd)
#define TJS     19           // LDS T row stride in u32 (16 pairs + 3 pad, odd)

#define X_WORDS (IN_DIM * XP)        // 10512 u32 per (hi|lo) plane
#define T_WORDS (16 * 16 * TJS)      // 4864  u32 per (hi|lo) plane
#define LDS_BYTES ((2 * X_WORDS + 2 * T_WORDS) * 4)  // 123008 B dynamic LDS

__device__ __forceinline__ uint16_t bf16_rne(float x) {
    uint32_t u = __float_as_uint(x);
    uint32_t r = u + 0x7FFFu + ((u >> 16) & 1u);
    return (uint16_t)(r >> 16);
}

__device__ __forceinline__ void split2(float x, uint16_t& hi, uint16_t& lo) {
    uint16_t h = bf16_rne(x);
    float fh = __uint_as_float(((uint32_t)h) << 16);
    lo = bf16_rne(x - fh);
    hi = h;
}

__device__ __forceinline__ uint32_t pack2(uint16_t lo16, uint16_t hi16) {
    return (uint32_t)lo16 | ((uint32_t)hi16 << 16);
}

__global__ __launch_bounds__(NTHREADS)
void conv2d_wmma_bf16x2_kernel(const float* __restrict__ x,
                               const float* __restrict__ wgt,
                               const float* __restrict__ bias,
                               float* __restrict__ out) {
    extern __shared__ uint32_t lds[];
    uint32_t* x_hi = lds;                     // [IN_DIM][XP]  packed bf16 pairs
    uint32_t* x_lo = x_hi + X_WORDS;
    uint32_t* t_hi = x_lo + X_WORDS;          // [16u][16j][TJS] pairs over k
    uint32_t* t_lo = t_hi + T_WORDS;

    const int tid = threadIdx.x;
    const int i0 = blockIdx.y * BT;
    const int j0 = blockIdx.x * BT;

    // ---- Phase 0: build Toeplitz weight planes (shared by whole grid) ----
    for (int idx = tid; idx < 16 * 16 * 16; idx += NTHREADS) {
        const int u  = idx >> 8;
        const int j  = (idx >> 4) & 15;
        const int kp = idx & 15;
        const int k0 = kp * 2, k1 = k0 + 1;
        const int d0 = k0 - j, d1 = k1 - j;
        float v0 = (d0 >= 0 && d0 < 16) ? wgt[u * 16 + d0] : 0.0f;
        float v1 = (d1 >= 0 && d1 < 16) ? wgt[u * 16 + d1] : 0.0f;
        uint16_t h0, l0, h1, l1;
        split2(v0, h0, l0);
        split2(v1, h1, l1);
        const int o = (u * 16 + j) * TJS + kp;
        t_hi[o] = pack2(h0, h1);
        t_lo[o] = pack2(l0, l1);
    }

    // ---- Phase 1: load 144x144 f32 halo patch, split -> bf16 hi/lo LDS ----
    // 36 float4 per row; clamp OOB addresses (results only feed masked outputs).
    for (int idx = tid; idx < IN_DIM * 36; idx += NTHREADS) {
        const int r  = idx / 36;
        const int c4 = idx % 36;
        int grow = i0 + r;      if (grow > HH - 1) grow = HH - 1;
        int gcol = j0 + c4 * 4; if (gcol > WW - 4) gcol = WW - 4;
        const float4 v = *reinterpret_cast<const float4*>(x + (size_t)grow * WW + gcol);
        uint16_t h0, l0, h1, l1, h2, l2, h3, l3;
        split2(v.x, h0, l0); split2(v.y, h1, l1);
        split2(v.z, h2, l2); split2(v.w, h3, l3);
        const int o = r * XP + c4 * 2;
        x_hi[o]     = pack2(h0, h1);
        x_hi[o + 1] = pack2(h2, h3);
        x_lo[o]     = pack2(l0, l1);
        x_lo[o + 1] = pack2(l2, l3);
    }

    __syncthreads();

    // ---- Phase 2: WMMA compute. Wave w owns output rows [w*16, w*16+16). ----
    const int wv   = tid >> 5;
    const int lane = tid & 31;
    const int m    = lane & 15;   // A row / B col / D col within tile
    const int kh   = lane >> 4;   // lane half selects K sub-range

    v8f acc[TJ];
#pragma unroll
    for (int jt = 0; jt < TJ; ++jt) acc[jt] = (v8f){0, 0, 0, 0, 0, 0, 0, 0};

    for (int u = 0; u < 16; ++u) {
        // B fragment (32x16): lanes 0-15 hold K=0..15, lanes 16-31 K=16..31.
        const uint32_t* tbh = t_hi + (u * 16 + m) * TJS + kh * 8;
        const uint32_t* tbl = t_lo + (u * 16 + m) * TJS + kh * 8;
        v8u bhu, blu;
#pragma unroll
        for (int p = 0; p < 8; ++p) { bhu[p] = tbh[p]; blu[p] = tbl[p]; }
        const v16bf Bh = __builtin_bit_cast(v16bf, bhu);
        const v16bf Bl = __builtin_bit_cast(v16bf, blu);

        // A fragment base: row = local output row + u (odd XP => conflict-free).
        const int arow = wv * 16 + m + u;
        const uint32_t* xrh = x_hi + arow * XP + kh * 8;
        const uint32_t* xrl = x_lo + arow * XP + kh * 8;

#pragma unroll
        for (int jt = 0; jt < TJ; ++jt) {
            v8u ahu, alu;
#pragma unroll
            for (int p = 0; p < 8; ++p) {
                // A-layout pair offset: vgpr p -> K pair p (+kh*8), matching B
                const int off = jt * 8 + p;
                ahu[p] = xrh[off];
                alu[p] = xrl[off];
            }
            const v16bf Ah = __builtin_bit_cast(v16bf, ahu);
            const v16bf Al = __builtin_bit_cast(v16bf, alu);
            acc[jt] = __builtin_amdgcn_wmma_f32_16x16x32_bf16(
                false, Ah, false, Bh, (short)0, acc[jt], false, false);
            acc[jt] = __builtin_amdgcn_wmma_f32_16x16x32_bf16(
                false, Ah, false, Bl, (short)0, acc[jt], false, false);
            acc[jt] = __builtin_amdgcn_wmma_f32_16x16x32_bf16(
                false, Al, false, Bh, (short)0, acc[jt], false, false);
        }
    }

    // ---- Phase 3: store. D layout: vgpr r -> M = r (lanes 0-15) / r+8. ----
    const float bval = bias[0];
    const int orow0 = i0 + wv * 16 + (kh ? 8 : 0);
#pragma unroll
    for (int jt = 0; jt < TJ; ++jt) {
        const int ocol = j0 + jt * 16 + m;
        if (ocol >= OW) continue;
#pragma unroll
        for (int r = 0; r < 8; ++r) {
            const int orow = orow0 + r;
            if (orow < OH)
                out[(size_t)orow * OW + ocol] = acc[jt][r] + bval;
        }
    }
}

extern "C" void kernel_launch(void* const* d_in, const int* in_sizes, int n_in,
                              void* d_out, int out_size, void* d_ws, size_t ws_size,
                              hipStream_t stream) {
    (void)in_sizes; (void)n_in; (void)out_size; (void)d_ws; (void)ws_size;
    const float* x    = (const float*)d_in[0];
    const float* wgt  = (const float*)d_in[1];
    const float* bias = (const float*)d_in[2];
    float* out        = (float*)d_out;

    static_assert(LDS_BYTES <= 160 * 1024, "LDS budget");
    (void)hipFuncSetAttribute((const void*)conv2d_wmma_bf16x2_kernel,
                              hipFuncAttributeMaxDynamicSharedMemorySize,
                              LDS_BYTES);

    dim3 grid((OW + BT - 1) / BT, (OH + BT - 1) / BT);
    conv2d_wmma_bf16x2_kernel<<<grid, NTHREADS, LDS_BYTES, stream>>>(x, wgt, bias, out);
}
